// DMPNNEncoder_61907658604848
// MI455X (gfx1250) — matrix operand order repaired and checked by reading, not compiled, not run on hardware
//
#include <hip/hip_runtime.h>
#include <stdint.h>

typedef __attribute__((ext_vector_type(16))) __bf16 v16bf;
typedef __attribute__((ext_vector_type(8)))  float  v8f;

static constexpr int MOLS  = 4096;
static constexpr int APM   = 24;
static constexpr int NATOM = MOLS * APM;       // 98304
static constexpr int EB    = MOLS * 2 * APM;   // 196608
static constexpr int H     = 300;
static constexpr int HP    = 320;              // 20 tiles of 16 -> 10 column pairs (2x2 blocking)
static constexpr int AD    = 133;
static constexpr int BD    = 14;
static constexpr int KI    = 160;              // 147 -> 160
static constexpr int KM    = 320;              // 300 -> 320
static constexpr int KA    = 448;              // 433 -> 448
static constexpr float EPSV = 1e-5f;

__device__ __forceinline__ unsigned short f2bf(float f) {
  unsigned int u = __float_as_uint(f);
  u += 0x7FFFu + ((u >> 16) & 1u);   // round-to-nearest-even
  return (unsigned short)(u >> 16);
}

union FragU { unsigned int u[8]; v16bf v; };

// ---------------- WMMA GEMM: Y[rows,HP] = X[rows,Kp](bf16) @ W[HP,Kp](bf16)^T ----------------
// 2x2 register-blocked: one wave computes a 32x32 output block (4 WMMA tiles),
// loading 2 A-frags + 2 B-frags per k-step -> 4 wmma : 4 frag loads (2x reuse).
// Lane layouts per CDNA5 ISA 7.12.2:
//   A (16x32 bf16): lane m=lane&15, half=lane>>4; vgpr i holds K pair
//       kA(i) = (i<4 ? 2i : 16+2(i-4)) + 8*half
//   B (32x16 bf16): lane n=lane&15; vgpr i holds K pair  kB(i) = kA(i)
//   D (16x16 f32):  vgpr i -> row (i + 8*half), col n
__global__ void k_gemm(const unsigned short* __restrict__ X, int ldx,
                       const unsigned short* __restrict__ W, int ldw,
                       float* __restrict__ Y,
                       int colPairs, int Kp, int nPairs)
{
  int wave = (blockIdx.x << 2) + (threadIdx.x >> 5);
  if (wave >= nPairs) return;                    // wave-uniform
  int lane = threadIdx.x & 31;
  int rp = wave / colPairs;
  int cp = wave - rp * colPairs;
  int n    = lane & 15;
  int half = lane >> 4;

  const unsigned short* ar0 = X + (size_t)(rp * 32 + n) * ldx;
  const unsigned short* ar1 = ar0 + (size_t)16 * ldx;
  const unsigned short* br0 = W + (size_t)(cp * 32 + n) * ldw;
  const unsigned short* br1 = br0 + (size_t)16 * ldw;

  v8f acc00 = {}, acc01 = {}, acc10 = {}, acc11 = {};
  for (int k0 = 0; k0 < Kp; k0 += 32) {
    FragU a0, a1, b0, b1;
#pragma unroll
    for (int i = 0; i < 8; ++i) {
      int ka = k0 + ((i < 4) ? (2 * i) : (16 + 2 * (i - 4))) + 8 * half;
      a0.u[i] = *(const unsigned int*)(ar0 + ka);
      a1.u[i] = *(const unsigned int*)(ar1 + ka);
      int kb = ka;
      b0.u[i] = *(const unsigned int*)(br0 + kb);
      b1.u[i] = *(const unsigned int*)(br1 + kb);
    }
    acc00 = __builtin_amdgcn_wmma_f32_16x16x32_bf16(false, a0.v, false, b0.v, (short)0, acc00, false, false);
    acc01 = __builtin_amdgcn_wmma_f32_16x16x32_bf16(false, a0.v, false, b1.v, (short)0, acc01, false, false);
    acc10 = __builtin_amdgcn_wmma_f32_16x16x32_bf16(false, a1.v, false, b0.v, (short)0, acc10, false, false);
    acc11 = __builtin_amdgcn_wmma_f32_16x16x32_bf16(false, a1.v, false, b1.v, (short)0, acc11, false, false);
  }

  float* yp = Y + (size_t)(rp * 32 + half * 8) * HP + cp * 32 + n;
#pragma unroll
  for (int i = 0; i < 8; ++i) {
    yp[(size_t)i * HP]            = acc00[i];
    yp[(size_t)i * HP + 16]       = acc01[i];
    yp[(size_t)(16 + i) * HP]      = acc10[i];
    yp[(size_t)(16 + i) * HP + 16] = acc11[i];
  }
}

// ---------------- weight conversion: f32 [rows,cols] -> bf16 [HP,Kp] zero-padded ----------------
__global__ void k_convw(const float* __restrict__ W, int rows, int cols,
                        unsigned short* __restrict__ Wb, int Kp, int total)
{
  int i = blockIdx.x * blockDim.x + threadIdx.x;
  if (i >= total) return;
  int r = i / Kp, c = i - r * Kp;
  float v = (r < rows && c < cols) ? W[(size_t)r * cols + c] : 0.f;
  Wb[i] = f2bf(v);
}

// ---------------- X_i = [bond_features | atom_features[bond_origins] | 0pad] (bf16) ----------------
__global__ void k_build_xi(const float* __restrict__ bondf, const float* __restrict__ atomf,
                           const int* __restrict__ bo, unsigned short* __restrict__ X)
{
  int e = blockIdx.x, c = threadIdx.x;          // blockDim == KI (160)
  float v = 0.f;
  if (c < BD)            v = bondf[(size_t)e * BD + c];
  else if (c < BD + AD)  v = atomf[(size_t)bo[e] * AD + (c - BD)];
  X[(size_t)e * KI + c] = f2bf(v);
}

__global__ void k_zero(float* __restrict__ p, int nv)
{
  int i = threadIdx.x;
  if (i < nv) p[i] = 0.f;
}

// per-column sum & sumsq, coalesced: threads = columns, blocks = row chunks
__global__ void k_colacc(const float* __restrict__ Y, int rowsPerBlock, float* __restrict__ acc)
{
  int c = threadIdx.x;                           // blockDim == 320
  if (c >= H) return;
  const float* p = Y + (size_t)blockIdx.x * rowsPerBlock * HP + c;
  float s = 0.f, s2 = 0.f;
  for (int r = 0; r < rowsPerBlock; ++r) {
    float v = p[(size_t)r * HP];
    s += v;
    s2 = fmaf(v, v, s2);
  }
  atomicAdd(&acc[c], s);
  atomicAdd(&acc[HP + c], s2);
}

// fold (g,b,mu,var) into per-column scale/shift
__global__ void k_bnfin(const float* __restrict__ acc, const float* __restrict__ g,
                        const float* __restrict__ b, float invRows,
                        float* __restrict__ scale, float* __restrict__ shift)
{
  int c = threadIdx.x;
  if (c >= H) return;
  float mean = acc[c] * invRows;
  float var  = acc[HP + c] * invRows - mean * mean;
  float rstd = rsqrtf(var + EPSV);
  float sc = g[c] * rstd;
  scale[c] = sc;
  shift[c] = b[c] - mean * sc;
}

__global__ void k_bnrelu_h0(const float* __restrict__ Y, const float* __restrict__ scale,
                            const float* __restrict__ shift,
                            float* __restrict__ h0, float* __restrict__ h)
{
  int r = blockIdx.x, c = threadIdx.x;
  if (c >= H) return;
  size_t idx = (size_t)r * HP + c;
  float v = fmaxf(fmaf(Y[idx], scale[c], shift[c]), 0.f);
  h0[idx] = v;
  h[idx]  = v;
}

__global__ void k_bnrelu_add(const float* __restrict__ Y, const float* __restrict__ h0,
                             const float* __restrict__ scale, const float* __restrict__ shift,
                             float* __restrict__ h)
{
  int r = blockIdx.x, c = threadIdx.x;
  if (c >= H) return;
  size_t idx = (size_t)r * HP + c;
  h[idx] = fmaxf(h0[idx] + fmaf(Y[idx], scale[c], shift[c]), 0.f);
}

__global__ void k_bnrelu_plain(const float* __restrict__ Y, const float* __restrict__ scale,
                               const float* __restrict__ shift, float* __restrict__ hv)
{
  int r = blockIdx.x, c = threadIdx.x;
  if (c >= H) return;
  size_t idx = (size_t)r * HP + c;
  hv[idx] = fmaxf(fmaf(Y[idx], scale[c], shift[c]), 0.f);
}

// m[e] = sum incoming (padded idx, 0 => zero row) - h[rev[e]]; emitted as bf16 GEMM input
__global__ void k_msg(const float* __restrict__ h, const int* __restrict__ bo,
                      const int* __restrict__ aibm, const int* __restrict__ rev,
                      unsigned short* __restrict__ Xm)
{
  int e = blockIdx.x, c = threadIdx.x;           // blockDim == KM (320)
  float v = 0.f;
  if (c < H) {
    int o  = bo[e];
    int i0 = aibm[2 * o], i1 = aibm[2 * o + 1];
    float s = 0.f;
    if (i0 > 0) s += h[(size_t)(i0 - 1) * HP + c];
    if (i1 > 0) s += h[(size_t)(i1 - 1) * HP + c];
    s -= h[(size_t)rev[e] * HP + c];
    v = s;
  }
  Xm[(size_t)e * KM + c] = f2bf(v);
}

// X_a = [atom_features | m_v | 0pad]; m_v[n] = h[2n] + h[2n+1] (ring: 2 out-bonds per atom)
__global__ void k_build_xa(const float* __restrict__ atomf, const float* __restrict__ h,
                           unsigned short* __restrict__ Xa)
{
  int nA = blockIdx.x, c = threadIdx.x;          // blockDim == KA (448)
  float v = 0.f;
  if (c < AD) {
    v = atomf[(size_t)nA * AD + c];
  } else if (c < AD + H) {
    int cc = c - AD;
    v = h[(size_t)(2 * nA) * HP + cc] + h[(size_t)(2 * nA + 1) * HP + cc];
  }
  Xa[(size_t)nA * KA + c] = f2bf(v);
}

// out[m, 0:300] = mean over 24 atoms of hv ; out[m, 300:500] = molecule_features
__global__ void k_out(const float* __restrict__ hv, const float* __restrict__ molf,
                      float* __restrict__ out)
{
  int m = blockIdx.x, c = threadIdx.x;           // blockDim == 512
  if (c < H) {
    const float* p = hv + (size_t)m * APM * HP + c;
    float s = 0.f;
#pragma unroll
    for (int v = 0; v < APM; ++v) s += p[(size_t)v * HP];
    out[(size_t)m * 500 + c] = s * (1.f / APM);
  } else if (c < H + 200) {
    out[(size_t)m * 500 + c] = molf[(size_t)m * 200 + (c - H)];
  }
}

extern "C" void kernel_launch(void* const* d_in, const int* in_sizes, int n_in,
                              void* d_out, int out_size, void* d_ws, size_t ws_size,
                              hipStream_t stream)
{
  (void)in_sizes; (void)n_in; (void)out_size; (void)ws_size;

  const float* atomf = (const float*)d_in[0];
  const float* bondf = (const float*)d_in[1];
  const int*   bo    = (const int*)d_in[2];
  const float* molf  = (const float*)d_in[3];
  const int*   aibm  = (const int*)d_in[4];
  const int*   rev   = (const int*)d_in[5];
  // d_in[6] mol_of_atom unused (atoms are contiguous per molecule)
  const float* W_i = (const float*)d_in[7];
  const float* g_i = (const float*)d_in[8];
  const float* b_i = (const float*)d_in[9];
  const float* W_m = (const float*)d_in[10];
  const float* g_m = (const float*)d_in[11];
  const float* b_m = (const float*)d_in[12];
  const float* W_a = (const float*)d_in[13];
  const float* g_a = (const float*)d_in[14];
  const float* b_a = (const float*)d_in[15];
  float* out = (float*)d_out;

  char* ws = (char*)d_ws;
  size_t off = 0;
  auto take = [&](size_t bytes) -> char* {
    char* p = ws + off;
    off = (off + bytes + 255) & ~(size_t)255;
    return p;
  };
  unsigned short* WiB = (unsigned short*)take((size_t)HP * KI * 2);
  unsigned short* WmB = (unsigned short*)take((size_t)HP * KM * 2);
  unsigned short* WaB = (unsigned short*)take((size_t)HP * KA * 2);
  float* acc   = (float*)take(2 * HP * 4);
  float* scale = (float*)take(HP * 4);
  float* shift = (float*)take(HP * 4);
  unsigned short* Xb = (unsigned short*)take((size_t)EB * KM * 2);  // shared X staging (max ld)
  float* Y  = (float*)take((size_t)EB * HP * 4);
  float* h0 = (float*)take((size_t)EB * HP * 4);
  float* h  = (float*)take((size_t)EB * HP * 4);

  // ---- weights to bf16, zero-padded ----
  k_convw<<<(HP * KI + 255) / 256, 256, 0, stream>>>(W_i, H, AD + BD, WiB, KI, HP * KI);
  k_convw<<<(HP * KM + 255) / 256, 256, 0, stream>>>(W_m, H, H,       WmB, KM, HP * KM);
  k_convw<<<(HP * KA + 255) / 256, 256, 0, stream>>>(W_a, H, AD + H,  WaB, KA, HP * KA);

  const int colP   = HP / 32;                     // 10 column pairs
  const int pairsE = (EB / 32) * colP;            // 61440
  const int pairsN = (NATOM / 32) * colP;         // 30720

  // ---- h0 = relu(BN(X_i @ W_i^T)) ----
  k_build_xi<<<EB, KI, 0, stream>>>(bondf, atomf, bo, Xb);
  k_gemm<<<pairsE / 4, 128, 0, stream>>>(Xb, KI, WiB, KI, Y, colP, KI, pairsE);
  k_zero<<<1, 2 * HP, 0, stream>>>(acc, 2 * HP);
  k_colacc<<<EB / 512, 320, 0, stream>>>(Y, 512, acc);
  k_bnfin<<<1, 320, 0, stream>>>(acc, g_i, b_i, 1.f / EB, scale, shift);
  k_bnrelu_h0<<<EB, 320, 0, stream>>>(Y, scale, shift, h0, h);

  // ---- message passing: h = relu(h0 + BN(m @ W_m^T)) ----
  for (int d = 0; d < 4; ++d) {
    k_msg<<<EB, KM, 0, stream>>>(h, bo, aibm, rev, Xb);
    k_gemm<<<pairsE / 4, 128, 0, stream>>>(Xb, KM, WmB, KM, Y, colP, KM, pairsE);
    k_zero<<<1, 2 * HP, 0, stream>>>(acc, 2 * HP);
    k_colacc<<<EB / 512, 320, 0, stream>>>(Y, 512, acc);
    k_bnfin<<<1, 320, 0, stream>>>(acc, g_m, b_m, 1.f / EB, scale, shift);
    k_bnrelu_add<<<EB, 320, 0, stream>>>(Y, h0, scale, shift, h);
  }

  // ---- atom readout: h_v = relu(BN(X_a @ W_a^T)) ----
  k_build_xa<<<NATOM, KA, 0, stream>>>(atomf, h, Xb);
  k_gemm<<<pairsN / 4, 128, 0, stream>>>(Xb, KA, WaB, KA, Y, colP, KA, pairsN);
  k_zero<<<1, 2 * HP, 0, stream>>>(acc, 2 * HP);
  k_colacc<<<NATOM / 512, 320, 0, stream>>>(Y, 512, acc);
  k_bnfin<<<1, 320, 0, stream>>>(acc, g_a, b_a, 1.f / NATOM, scale, shift);
  k_bnrelu_plain<<<NATOM, 320, 0, stream>>>(Y, scale, shift, h0);   // h_v stored in h0 buffer

  // ---- molecule mean + concat molecule_features ----
  k_out<<<MOLS, 512, 0, stream>>>(h0, molf, out);
}
